// PairStack_31645319037645
// MI455X (gfx1250) — compile-verified, matrix operand order checked by reading, not run on hardware
//
#include <hip/hip_runtime.h>
#include <math.h>

// ---------------------------------------------------------------------------
// PairStack (triangle-mult x2, triangle-attn x2, transition) for gfx1250.
// All contractions via v_wmma_f32_16x16x32_bf16 (f32 accum); bf16 intermediates
// halve HBM traffic against the 23.3 TB/s roofline. WMMA fragments are fetched
// as ds_load_b128 pairs (ISA 7.12.2: two contiguous 8-elem runs per lane);
// B operands are kept column-major (B^T) so A and B tiles stage and gather
// with b128 ops. Bias/mask terms are hoisted out of the WMMA loops into the
// vectorized softmax. Workspace: ~132 MB of d_ws, regions reused.
// ---------------------------------------------------------------------------

typedef __bf16 bf16;
typedef __attribute__((ext_vector_type(16))) __bf16 v16bf;
typedef __attribute__((ext_vector_type(8)))  float  v8f;
typedef __attribute__((ext_vector_type(4)))  float  f32x4;
typedef __attribute__((ext_vector_type(4)))  unsigned int u32x4;
typedef __attribute__((ext_vector_type(2)))  unsigned int u32x2;

#define NSEQ  256
#define NSQ2  65536
#define DPAIR 128

#define TMT 64          // GEMM tile M
#define TNT 64          // GEMM tile N
#define TKT 32          // GEMM tile K (one bf16 WMMA K-step)
#define TSTR 40         // LDS tile line stride (32 elems + 8 pad), 80 B

enum { EPI_F32 = 0, EPI_BF16 = 1, EPI_SIG = 2, EPI_RELU = 3,
       EPI_RESGATE = 4, EPI_RESADD = 5, EPI_TRIPACK = 6, EPI_ATTNV = 7 };

__device__ __forceinline__ v8f wmma_bf16(v16bf a, v16bf b, v8f c) {
  return __builtin_amdgcn_wmma_f32_16x16x32_bf16(false, a, false, b,
                                                 (short)0, c, false, false);
}

// One WMMA operand fragment (A or column-major B) from LDS.
// Lane owns line (lineBase + lane&15); its 16 bf16 elems are two contiguous
// 8-element runs at k = 8h.. and k = 16+8h.. -> two ds_load_b128.
__device__ __forceinline__ v16bf frag16(const bf16* s, int lineBase, int stride, int kOff) {
  const int lane = threadIdx.x & 31;
  const bf16* p = s + (lineBase + (lane & 15)) * stride + kOff + ((lane >> 4) * 8);
  v16bf f;
  *(u32x4*)(&f)       = *(const u32x4*)(p);
  *((u32x4*)(&f) + 1) = *(const u32x4*)(p + 16);
  return f;
}

__device__ __forceinline__ float sigmoidf_(float x) { return 1.f / (1.f + __expf(-x)); }

// ---------------------------------------------------------------------------
// Generic WMMA GEMM: C[M,N] = A[M,K](bf16,row-major) * Bt[N,K](bf16,B^T).
// 128 threads = 4 waves; each wave owns a 32x32 C tile (4 accumulators).
// DUAL adds a second B^T sharing the A tile (gated dual projection).
// Epilogue dispatch is hoisted outside the unrolled store loops.
// ---------------------------------------------------------------------------
template<bool DUAL>
__global__ void __launch_bounds__(128) wmma_gemm(
    const bf16* __restrict__ A, const bf16* __restrict__ B, const bf16* __restrict__ B2,
    const float* __restrict__ bias, const float* __restrict__ bias2,
    const float* __restrict__ mask, const bf16* __restrict__ gate,
    const float* __restrict__ resid,
    float* __restrict__ outF, bf16* __restrict__ outB, bf16* __restrict__ outB2,
    int M, int N, int K, int lda, int ldb,
    long orows, long ocols, long batchA, long batchB, long batchO,
    float scale, int epi, int triMode, int transStore)
{
  __shared__ bf16 As[TMT * TSTR];    // 64 rows x 32 k
  __shared__ bf16 Bs[TNT * TSTR];    // 64 cols x 32 k (column-major tile)
  __shared__ bf16 Bs2[TNT * TSTR];

  const int tid  = threadIdx.x;
  const int lane = tid & 31;
  const int wave = tid >> 5;
  const int wm = wave >> 1, wn = wave & 1;
  const int m0 = blockIdx.y * TMT;
  const int n0 = blockIdx.x * TNT;
  const long zb = blockIdx.z;
  const bf16* Ab  = A + zb * batchA;
  const bf16* Bb  = B + zb * batchB;
  const bf16* B2b = DUAL ? (B2 + zb * batchB) : B2;

  v8f acc[2][2]  = {};
  v8f acc2[2][2] = {};

  for (int k0 = 0; k0 < K; k0 += TKT) {
#pragma unroll
    for (int it = 0; it < 2; ++it) {
      int idx = tid + it * 128;          // 256 vec8 chunks per tile
      int row = idx >> 2;
      int ck  = (idx & 3) * 8;
      *(u32x4*)(&As[row * TSTR + ck]) =
          *(const u32x4*)(Ab + (long)(m0 + row) * lda + k0 + ck);
      u32x4 bv = {};
      if (n0 + row < N)
        bv = *(const u32x4*)(Bb + (long)(n0 + row) * ldb + k0 + ck);
      *(u32x4*)(&Bs[row * TSTR + ck]) = bv;
      if (DUAL) {
        u32x4 bv2 = {};
        if (n0 + row < N)
          bv2 = *(const u32x4*)(B2b + (long)(n0 + row) * ldb + k0 + ck);
        *(u32x4*)(&Bs2[row * TSTR + ck]) = bv2;
      }
    }
    if (k0 + TKT < K)  // global_prefetch_b8 of next A K-tile
      __builtin_prefetch(Ab + (long)(m0 + (tid >> 1)) * lda + k0 + TKT, 0, 1);
    __syncthreads();

    v16bf af0 = frag16(As, 32 * wm,      TSTR, 0);
    v16bf af1 = frag16(As, 32 * wm + 16, TSTR, 0);
#pragma unroll
    for (int ni = 0; ni < 2; ++ni) {
      v16bf bfr = frag16(Bs, 32 * wn + 16 * ni, TSTR, 0);
      acc[0][ni] = wmma_bf16(af0, bfr, acc[0][ni]);
      acc[1][ni] = wmma_bf16(af1, bfr, acc[1][ni]);
    }
    if (DUAL) {
#pragma unroll
      for (int ni = 0; ni < 2; ++ni) {
        v16bf bfr = frag16(Bs2, 32 * wn + 16 * ni, TSTR, 0);
        acc2[0][ni] = wmma_bf16(af0, bfr, acc2[0][ni]);
        acc2[1][ni] = wmma_bf16(af1, bfr, acc2[1][ni]);
      }
    }
    __syncthreads();
  }

  // ---- epilogue: C element (row = e + 8*(lane>>4), col = lane&15) ----
  const int h2 = lane >> 4, cl = lane & 15;
  auto forEach = [&](auto&& fn) {
#pragma unroll
    for (int mi = 0; mi < 2; ++mi)
#pragma unroll
      for (int ni = 0; ni < 2; ++ni)
#pragma unroll
        for (int e = 0; e < 8; ++e) {
          int gr = m0 + 32 * wm + 16 * mi + 8 * h2 + e;
          int gc = n0 + 32 * wn + 16 * ni + cl;
          if (gc < N && gr < M)
            fn(gr, gc, acc[mi][ni][e] * scale, acc2[mi][ni][e]);
        }
  };
  switch (epi) {
    case EPI_F32:
      forEach([&](int gr, int gc, float v, float) {
        float bv = bias ? bias[gc] : 0.f;
        outF[(long)gr * orows + (long)gc * ocols + zb * batchO] = v + bv;
      });
      break;
    case EPI_BF16:
      forEach([&](int gr, int gc, float v, float) {
        float bv = bias ? bias[gc] : 0.f;
        outB[(long)gr * orows + (long)gc * ocols + zb * batchO] = (bf16)(v + bv);
      });
      break;
    case EPI_SIG:
      forEach([&](int gr, int gc, float v, float) {
        outB[(long)gr * orows + (long)gc * ocols] = (bf16)sigmoidf_(v + bias[gc]);
      });
      break;
    case EPI_RELU:
      forEach([&](int gr, int gc, float v, float) {
        outB[(long)gr * orows + (long)gc * ocols] = (bf16)fmaxf(v + bias[gc], 0.f);
      });
      break;
    case EPI_RESGATE:   // z_raw + g * (acc + ob)
      forEach([&](int gr, int gc, float v, float) {
        long idx = (long)gr * DPAIR + gc;
        outF[idx] = resid[idx] + (float)gate[idx] * (v + bias[gc]);
      });
      break;
    case EPI_RESADD:    // (optionally transposed) z_raw + acc + ob
      forEach([&](int gr, int gc, float v, float) {
        long orow = transStore ? (long)(gr & 255) * NSEQ + (gr >> 8) : (long)gr;
        long idx = orow * DPAIR + gc;
        float bv = bias ? bias[gc] : 0.f;
        outF[idx] = resid[idx] + v + bv;
      });
      break;
    case EPI_TRIPACK:   // (acc1+b1)*mask*sigmoid(acc2+b2) -> einsum A / B^T pack
      forEach([&](int gr, int gc, float v, float v2) {
        float p = (v + bias[gc]) * mask[gr];
        p *= sigmoidf_(v2 + bias2[gc]);
        long rr = (triMode == 0) ? (long)gr
                                 : ((long)(gr & 255) * NSEQ + (gr >> 8));
        bf16* dst = ((gc >> 7) == 0) ? outB : outB2;
        dst[(long)(gc & 127) * NSQ2 + rr] = (bf16)p;
      });
      break;
    case EPI_ATTNV:     // v-proj written transposed: vT[(i*128+gc)*256 + j]
      forEach([&](int gr, int gc, float v, float) {
        long idx = ((long)(gr >> 8) * DPAIR + gc) * NSEQ + (gr & 255);
        outB[idx] = (bf16)v;
      });
      break;
  }
}

// ---------------------------------------------------------------------------
// LayerNorm over last dim (128) -> bf16. One wave per row, vectorized IO,
// shfl_xor reductions (wave32).
// ---------------------------------------------------------------------------
__global__ void __launch_bounds__(256) ln_kernel(
    const float* __restrict__ inF, const bf16* __restrict__ inB,
    const float* __restrict__ s, const float* __restrict__ bsc,
    bf16* __restrict__ out, int transRead)
{
  int row  = blockIdx.x * 8 + (threadIdx.x >> 5);
  int lane = threadIdx.x & 31;
  long src;
  if (transRead) { int u = row >> 8, vv = row & 255; src = ((long)vv * NSEQ + u) * DPAIR; }
  else           { src = (long)row * DPAIR; }
  const int j0 = lane * 4;
  f32x4 x;
  if (inF) {
    x = *(const f32x4*)(inF + src + j0);
  } else {
    u32x2 raw = *(const u32x2*)(inB + src + j0);
    const bf16* rb = (const bf16*)&raw;
    x[0] = (float)rb[0]; x[1] = (float)rb[1]; x[2] = (float)rb[2]; x[3] = (float)rb[3];
  }
  float sum = x[0] + x[1] + x[2] + x[3];
#pragma unroll
  for (int off = 16; off > 0; off >>= 1) sum += __shfl_xor(sum, off, 32);
  float mean = sum * (1.f / 128.f);
  float vs = 0.f;
#pragma unroll
  for (int t = 0; t < 4; ++t) { float d = x[t] - mean; vs += d * d; }
#pragma unroll
  for (int off = 16; off > 0; off >>= 1) vs += __shfl_xor(vs, off, 32);
  float inv = rsqrtf(vs * (1.f / 128.f) + 1e-5f);
  f32x4 sv = *(const f32x4*)(s + j0);
  f32x4 bv = *(const f32x4*)(bsc + j0);
  union { u32x2 v; bf16 h[4]; } o4;
#pragma unroll
  for (int t = 0; t < 4; ++t) o4.h[t] = (bf16)((x[t] - mean) * inv * sv[t] + bv[t]);
  *(u32x2*)(out + (long)row * DPAIR + j0) = o4.v;
}

// ---------------------------------------------------------------------------
// Triangle attention core: one block per (i, h, 64-row q tile).
// Q / K / V^T / S / P resident in LDS (~143 KB < 320 KB WGP budget).
// S = QK^T and O = PV via WMMA. Bias (head-major) and mask row are applied in
// the vectorized softmax, not in the WMMA fragment loop.
// ---------------------------------------------------------------------------
#define SSTR 260
#define PSTR 264
#define VTSTR 264

__global__ void __launch_bounds__(128) attn_kernel(
    const bf16* __restrict__ q, const bf16* __restrict__ k,
    const bf16* __restrict__ vT, const float* __restrict__ biasT,
    const float* __restrict__ msel, bf16* __restrict__ o)
{
  __shared__ bf16  Qs[64 * TSTR];     // [64 q rows][32 ca]
  __shared__ bf16  Ks[256 * TSTR];    // [256 j][32 ca] == column-major K^T
  __shared__ bf16  VTs[32 * VTSTR];   // [32 ca][256 j] == column-major V
  __shared__ float Ss[64 * SSTR];     // logits / exp
  __shared__ bf16  Ps[64 * PSTR];     // softmax probs (A for PV)
  __shared__ float red[128];

  const int tid  = threadIdx.x;
  const int lane = tid & 31;
  const int wave = tid >> 5;
  const int q0 = blockIdx.x * 64;
  const int hh = blockIdx.y;
  const int ii = blockIdx.z;
  const long rowbase = (long)ii * NSEQ;

  // Q tile (b128 copies)
#pragma unroll
  for (int it = 0; it < 2; ++it) {
    int idx = tid + it * 128;
    int r = idx >> 2, cg = (idx & 3) * 8;
    *(u32x4*)(&Qs[r * TSTR + cg]) =
        *(const u32x4*)(q + (rowbase + q0 + r) * DPAIR + hh * 32 + cg);
  }
  // K: row-major K == column-major K^T -> direct copy
#pragma unroll
  for (int it = 0; it < 8; ++it) {
    int idx = tid + it * 128;
    int j = idx >> 2, cg = (idx & 3) * 8;
    *(u32x4*)(&Ks[j * TSTR + cg]) =
        *(const u32x4*)(k + (rowbase + j) * DPAIR + hh * 32 + cg);
  }
  // V^T: pre-transposed in global (EPI_ATTNV) -> direct copy
#pragma unroll
  for (int it = 0; it < 2; ++it) {
    int idx = tid + it * 128;            // 32 lines x 8 chunks = 256
    int ca = idx >> 3, jg = (idx & 7) * 32;
#pragma unroll
    for (int c4 = 0; c4 < 4; ++c4)
      *(u32x4*)(&VTs[ca * VTSTR + jg + c4 * 8]) =
          *(const u32x4*)(vT + ((long)ii * DPAIR + hh * 32 + ca) * NSEQ + jg + c4 * 8);
  }
  __syncthreads();

  // S = Q K^T  (each wave: 16 q rows x 256 cols, K=32 single step)
  {
    v16bf a = frag16(Qs, 16 * wave, TSTR, 0);
    const int h2 = lane >> 4, cl = lane & 15;
    for (int nt = 0; nt < 16; ++nt) {
      v8f accS = {};
      v16bf b = frag16(Ks, 16 * nt, TSTR, 0);
      accS = wmma_bf16(a, b, accS);
#pragma unroll
      for (int e = 0; e < 8; ++e)
        Ss[(16 * wave + 8 * h2 + e) * SSTR + 16 * nt + cl] = accS[e];
    }
  }
  __syncthreads();

  // softmax: thread = (row = tid>>1, half = tid&1); bias+mask folded into the
  // vectorized max pass; LDS combine across the two halves.
  {
    int r = tid >> 1, hseg = tid & 1;
    float* rowp = &Ss[r * SSTR + hseg * 128];
    const float* bT = biasT + (long)hh * NSQ2 + (long)(q0 + r) * NSEQ + hseg * 128;
    const float* mr = msel + (long)ii * NSEQ + hseg * 128;
    float m = -3.4e38f;
    for (int j = 0; j < 128; j += 4) {
      f32x4 xx = *(f32x4*)(rowp + j);
      f32x4 bb = *(const f32x4*)(bT + j);
      f32x4 mm = *(const f32x4*)(mr + j);
#pragma unroll
      for (int t = 0; t < 4; ++t) xx[t] += bb[t] + 1e9f * (mm[t] - 1.f);
      *(f32x4*)(rowp + j) = xx;
      m = fmaxf(m, fmaxf(fmaxf(xx[0], xx[1]), fmaxf(xx[2], xx[3])));
    }
    red[tid] = m;
    __syncthreads();
    m = fmaxf(red[r * 2], red[r * 2 + 1]);
    float ssum = 0.f;
    for (int j = 0; j < 128; j += 4) {
      f32x4 xx = *(const f32x4*)(rowp + j);
      f32x4 ee;
#pragma unroll
      for (int t = 0; t < 4; ++t) ee[t] = __expf(xx[t] - m);
      *(f32x4*)(rowp + j) = ee;
      ssum += ee[0] + ee[1] + ee[2] + ee[3];
    }
    __syncthreads();
    red[tid] = ssum;
    __syncthreads();
    float inv = 1.f / (red[r * 2] + red[r * 2 + 1]);
    bf16* prow = &Ps[r * PSTR + hseg * 128];
    for (int j = 0; j < 128; j += 8) {
      union { u32x4 v; bf16 h[8]; } pk;
#pragma unroll
      for (int t = 0; t < 8; ++t) pk.h[t] = (bf16)(rowp[j + t] * inv);
      *(u32x4*)(prow + j) = pk.v;
    }
  }
  __syncthreads();

  // O = P V (each wave: 16 rows x 32 cols, K=256 over 8 steps)
  {
    v8f accO[2] = {};
    for (int kt = 0; kt < 8; ++kt) {
      v16bf a = frag16(Ps, 16 * wave, PSTR, 32 * kt);
#pragma unroll
      for (int ni = 0; ni < 2; ++ni) {
        v16bf b = frag16(VTs, 16 * ni, VTSTR, 32 * kt);
        accO[ni] = wmma_bf16(a, b, accO[ni]);
      }
    }
    const int h2 = lane >> 4, cl = lane & 15;
#pragma unroll
    for (int ni = 0; ni < 2; ++ni)
#pragma unroll
      for (int e = 0; e < 8; ++e) {
        int lr = 16 * wave + 8 * h2 + e;
        int c  = 16 * ni + cl;
        o[(rowbase + q0 + lr) * DPAIR + hh * 32 + c] = (bf16)accO[ni][e];
      }
  }
}

// ---------------------------------------------------------------------------
// Small elementwise kernels
// ---------------------------------------------------------------------------
// f32 [K,N] -> bf16 transposed [N,K] (weights stored as B^T)
__global__ void __launch_bounds__(256) f32_to_bf16_t(const float* __restrict__ in,
                                                     bf16* __restrict__ out,
                                                     int K, int N) {
  int idx = blockIdx.x * 256 + threadIdx.x;
  if (idx < K * N) {
    int kk = idx / N, nn = idx % N;
    out[nn * K + kk] = (bf16)in[idx];
  }
}

// mask [256,256] -> maskT (for ending-node attention rows)
__global__ void __launch_bounds__(256) transpose_mask(const float* __restrict__ in,
                                                      float* __restrict__ out) {
  int x = blockIdx.x * 256 + threadIdx.x;      // 65536 elements
  out[(x & 255) * NSEQ + (x >> 8)] = in[x];
}

__global__ void __launch_bounds__(256) mul_bf16_k(bf16* __restrict__ a,
                                                  const bf16* __restrict__ b) {
  int i = (blockIdx.x * 256 + threadIdx.x) * 8;
  u32x4 av = *(const u32x4*)(a + i);
  u32x4 bv = *(const u32x4*)(b + i);
  const bf16* ah = (const bf16*)&av;
  const bf16* bh = (const bf16*)&bv;
  union { u32x4 v; bf16 h[8]; } o;
#pragma unroll
  for (int e = 0; e < 8; ++e) o.h[e] = (bf16)((float)ah[e] * (float)bh[e]);
  *(u32x4*)(a + i) = o.v;
}

// ---------------------------------------------------------------------------
// Host orchestration
// ---------------------------------------------------------------------------
extern "C" void kernel_launch(void* const* d_in, const int* in_sizes, int n_in,
                              void* d_out, int out_size, void* d_ws, size_t ws_size,
                              hipStream_t stream)
{
  (void)in_sizes; (void)n_in; (void)out_size; (void)ws_size;
  int ii = 0;
  auto nf = [&]() { return (const float*)d_in[ii++]; };

  const float* pair  = nf();
  const float* pmask = nf();

  struct TMu { const float *ln1s,*ln1b,*lrpw,*lrpb,*lrgw,*lrgb,*ogw,*ogb,*ln2s,*ln2b,*opw,*ob; };
  struct TAt { const float *ln1s,*ln1b,*bw,*qw,*kw,*vw,*gw,*gb,*ow,*ob; };
  TMu tmo{nf(),nf(),nf(),nf(),nf(),nf(),nf(),nf(),nf(),nf(),nf(),nf()};
  TMu tmi{nf(),nf(),nf(),nf(),nf(),nf(),nf(),nf(),nf(),nf(),nf(),nf()};
  TAt tas{nf(),nf(),nf(),nf(),nf(),nf(),nf(),nf(),nf(),nf()};
  TAt tae{nf(),nf(),nf(),nf(),nf(),nf(),nf(),nf(),nf(),nf()};
  const float *trs = nf(), *trb = nf(), *trw1 = nf(), *trb1 = nf(),
              *trw2 = nf(), *trb2 = nf();

  // workspace layout (d_ws assumed >= ~132 MB)
  char* ws = (char*)d_ws;
  float* zcur   = (float*)(ws + 0);                 // 32 MB running pair (f32)
  bf16*  zln    = (bf16*)(ws + (32ll  << 20));      // 16 MB LN(z) bf16
  bf16*  bufC   = (bf16*)(ws + (48ll  << 20));      // 16 MB Lt / q / transition-h
  bf16*  bufD   = (bf16*)(ws + (64ll  << 20));      // 16 MB Rt^T / k
  bf16*  bufE   = (bf16*)(ws + (80ll  << 20));      // 16 MB ab / v^T
  bf16*  bufF   = (bf16*)(ws + (96ll  << 20));      // 16 MB abln / o
  bf16*  bufG   = (bf16*)(ws + (112ll << 20));      // 16 MB gate
  float* biasbf = (float*)(ws + (128ll << 20));     // 1 MB attn bias, head-major [4][N2]
  float* maskT  = (float*)(ws + (129ll << 20));     // 256 KB transposed mask
  bf16*  wpool  = (bf16*)(ws + (130ll << 20));      // ~1 MB bf16 weights (B^T)
  size_t wcur = 0;

  auto cvt_t = [&](const float* src, int K, int N) -> const bf16* {
    bf16* dst = wpool + wcur;
    int n = K * N;
    wcur += (size_t)((n + 127) & ~127);
    f32_to_bf16_t<<<dim3((n + 255) / 256), dim3(256), 0, stream>>>(src, dst, K, N);
    return dst;
  };
  auto ln = [&](const float* inF, const bf16* inB, const float* s, const float* b,
                bf16* out, int trans) {
    ln_kernel<<<dim3(NSQ2 / 8), dim3(256), 0, stream>>>(inF, inB, s, b, out, trans);
  };
  auto gemm = [&](const bf16* A, const bf16* Bt, const float* bias,
                  float* oF, bf16* oB, int M, int N, int K, int lda, int ldb,
                  long orows, long ocols, float scale, int epi,
                  const bf16* gate, const float* resid, int transStore,
                  int batch, long bA, long bB, long bO) {
    dim3 g((N + TNT - 1) / TNT, M / TMT, batch);
    wmma_gemm<false><<<g, dim3(128), 0, stream>>>(
        A, Bt, (const bf16*)nullptr, bias, (const float*)nullptr,
        (const float*)nullptr, gate, resid, oF, oB, (bf16*)nullptr,
        M, N, K, lda, ldb, orows, ocols, bA, bB, bO, scale, epi, 0, transStore);
  };

  // z = pair (mutable running copy); maskT precomputed once
  hipMemcpyAsync(zcur, pair, (size_t)NSQ2 * DPAIR * sizeof(float),
                 hipMemcpyDeviceToDevice, stream);
  transpose_mask<<<dim3(NSQ2 / 256), dim3(256), 0, stream>>>(pmask, maskT);

  auto tri_mul = [&](const TMu& p, int mode) {   // mode 0=outgoing, 1=incoming
    const bf16* lrpw = cvt_t(p.lrpw, DPAIR, 2 * DPAIR);
    const bf16* lrgw = cvt_t(p.lrgw, DPAIR, 2 * DPAIR);
    const bf16* ogw  = cvt_t(p.ogw,  DPAIR, DPAIR);
    const bf16* opw  = cvt_t(p.opw,  DPAIR, DPAIR);
    ln(zcur, nullptr, p.ln1s, p.ln1b, zln, 0);
    {  // fused dual gated projection, packed straight into A / B^T einsum layouts
      dim3 g(256 / TNT, NSQ2 / TMT, 1);
      wmma_gemm<true><<<g, dim3(128), 0, stream>>>(
          zln, lrpw, lrgw, p.lrpb, p.lrgb, pmask, (const bf16*)nullptr,
          (const float*)nullptr, (float*)nullptr, bufC, bufD,
          NSQ2, 256, DPAIR, DPAIR, DPAIR, 0, 0, 0, 0, 0,
          1.f, EPI_TRIPACK, mode, 0);
    }
    gemm(zln, ogw, p.ogb, nullptr, bufG, NSQ2, DPAIR, DPAIR, DPAIR, DPAIR,
         DPAIR, 1, 1.f, EPI_SIG, nullptr, nullptr, 0, 1, 0, 0, 0);
    // triangle einsum: 128 independent 256x256x256 bf16 GEMMs (batched over c)
    gemm(bufC, bufD, nullptr, nullptr, bufE, NSEQ, NSEQ, NSEQ, NSEQ, NSEQ,
         (long)NSEQ * DPAIR, DPAIR, 1.f, EPI_BF16, nullptr, nullptr, 0,
         DPAIR, (long)NSQ2, (long)NSQ2, 1);
    ln(nullptr, bufE, p.ln2s, p.ln2b, bufF, 0);
    gemm(bufF, opw, p.ob, zcur, nullptr, NSQ2, DPAIR, DPAIR, DPAIR, DPAIR,
         DPAIR, 1, 1.f, EPI_RESGATE, bufG, zcur, 0, 1, 0, 0, 0);
  };

  auto tri_attn = [&](const TAt& p, int starting) {
    const bf16* bw = cvt_t(p.bw, DPAIR, 4);
    const bf16* qw = cvt_t(p.qw, DPAIR, DPAIR);
    const bf16* kw = cvt_t(p.kw, DPAIR, DPAIR);
    const bf16* vw = cvt_t(p.vw, DPAIR, DPAIR);
    const bf16* gw = cvt_t(p.gw, DPAIR, DPAIR);
    const bf16* ow = cvt_t(p.ow, DPAIR, DPAIR);
    ln(zcur, nullptr, p.ln1s, p.ln1b, zln, starting ? 0 : 1);
    gemm(zln, qw, nullptr, nullptr, bufC, NSQ2, DPAIR, DPAIR, DPAIR, DPAIR,
         DPAIR, 1, 0.17677669529663689f /* 32^-0.5 */, EPI_BF16,
         nullptr, nullptr, 0, 1, 0, 0, 0);
    gemm(zln, kw, nullptr, nullptr, bufD, NSQ2, DPAIR, DPAIR, DPAIR, DPAIR,
         DPAIR, 1, 1.f, EPI_BF16, nullptr, nullptr, 0, 1, 0, 0, 0);
    gemm(zln, vw, nullptr, nullptr, bufE, NSQ2, DPAIR, DPAIR, DPAIR, DPAIR,
         0, 0, 1.f, EPI_ATTNV, nullptr, nullptr, 0, 1, 0, 0, 0);
    // triangle bias, head-major: biasT[h][q*256+j]
    gemm(zln, bw, nullptr, biasbf, nullptr, NSQ2, 4, DPAIR, DPAIR, DPAIR,
         1, NSQ2, 1.f, EPI_F32, nullptr, nullptr, 0, 1, 0, 0, 0);
    attn_kernel<<<dim3(NSEQ / 64, 4, NSEQ), dim3(128), 0, stream>>>(
        bufC, bufD, bufE, biasbf, starting ? pmask : maskT, bufF);
    gemm(zln, gw, p.gb, nullptr, bufG, NSQ2, DPAIR, DPAIR, DPAIR, DPAIR,
         DPAIR, 1, 1.f, EPI_SIG, nullptr, nullptr, 0, 1, 0, 0, 0);
    mul_bf16_k<<<dim3((NSQ2 * DPAIR) / (256 * 8)), dim3(256), 0, stream>>>(bufF, bufG);
    gemm(bufF, ow, p.ob, zcur, nullptr, NSQ2, DPAIR, DPAIR, DPAIR, DPAIR,
         DPAIR, 1, 1.f, EPI_RESADD, nullptr, zcur, starting ? 0 : 1, 1, 0, 0, 0);
  };

  tri_mul(tmo, 0);
  tri_mul(tmi, 1);
  tri_attn(tas, 1);
  tri_attn(tae, 0);

  // transition
  {
    const bf16* w1 = cvt_t(trw1, DPAIR, 4 * DPAIR);
    const bf16* w2 = cvt_t(trw2, 4 * DPAIR, DPAIR);
    ln(zcur, nullptr, trs, trb, zln, 0);
    bf16* hbuf = bufC;  // spans bufC..bufF (64 MB = N2 x 512 bf16)
    gemm(zln, w1, trb1, nullptr, hbuf, NSQ2, 4 * DPAIR, DPAIR, DPAIR, DPAIR,
         4 * DPAIR, 1, 1.f, EPI_RELU, nullptr, nullptr, 0, 1, 0, 0, 0);
    gemm(hbuf, w2, trb2, (float*)d_out, nullptr, NSQ2, DPAIR, 4 * DPAIR,
         4 * DPAIR, 4 * DPAIR, DPAIR, 1, 1.f, EPI_RESADD, nullptr, zcur, 0,
         1, 0, 0, 0);
  }
}
